// GIN_13400297964012
// MI455X (gfx1250) — matrix operand order repaired and checked by reading, not compiled, not run on hardware
//
#include <hip/hip_runtime.h>
#include <hip/hip_bf16.h>

// ---------------------------------------------------------------------------
// Types for CDNA5 WMMA (wave32): v_wmma_f32_16x16x32_bf16
// ---------------------------------------------------------------------------
typedef __attribute__((ext_vector_type(16))) __bf16 v16bf;
typedef __attribute__((ext_vector_type(8)))  __bf16 v8bf;
typedef __attribute__((ext_vector_type(8)))  float  v8f;

__device__ __forceinline__ unsigned short f32_to_bf16_rne(float f) {
    unsigned int u = __float_as_uint(f);
    unsigned int r = (u + 0x7FFFu + ((u >> 16) & 1u)) >> 16;
    return (unsigned short)r;
}

union Pack8 {
    unsigned short s[8];
    uint4 u;
};

__device__ __forceinline__ Pack8 pack_f8(float4 v0, float4 v1) {
    Pack8 pk;
    pk.s[0] = f32_to_bf16_rne(v0.x); pk.s[1] = f32_to_bf16_rne(v0.y);
    pk.s[2] = f32_to_bf16_rne(v0.z); pk.s[3] = f32_to_bf16_rne(v0.w);
    pk.s[4] = f32_to_bf16_rne(v1.x); pk.s[5] = f32_to_bf16_rne(v1.y);
    pk.s[6] = f32_to_bf16_rne(v1.z); pk.s[7] = f32_to_bf16_rne(v1.w);
    return pk;
}

// Build a 16-element bf16 fragment from two aligned 8-element LDS runs.
__device__ __forceinline__ v16bf make_frag(const unsigned short* p0,
                                           const unsigned short* p1) {
    v8bf lo = *reinterpret_cast<const v8bf*>(p0);
    v8bf hi = *reinterpret_cast<const v8bf*>(p1);
    v16bf r;
#pragma unroll
    for (int e = 0; e < 8; ++e) { r[e] = lo[e]; r[e + 8] = hi[e]; }
    return r;
}

__device__ __forceinline__ v8f wmma_bf16(v16bf a, v16bf b, v8f c) {
    return __builtin_amdgcn_wmma_f32_16x16x32_bf16(
               false, a, false, b, (short)0, c, false, false);
}

// ---------------------------------------------------------------------------
// One-time weight prep: Wt[n*K + k] = bf16(W[k*N + n])   (tiny: <= 64K elems)
// ---------------------------------------------------------------------------
__global__ __launch_bounds__(256) void transpose_w_bf16(
    const float* __restrict__ W, unsigned short* __restrict__ Wt, int K, int N)
{
    int i = blockIdx.x * blockDim.x + threadIdx.x;
    if (i < K * N) {
        int n = i / K, k = i % K;
        Wt[i] = f32_to_bf16_rne(W[(size_t)k * N + n]);
    }
}

// ---------------------------------------------------------------------------
// GEMM: C[M,N] = act(A[M,K] @ W[K,N] + bias[N])
//   A fp32 row-major; W pre-transposed bf16 (Wt[n][k]); C fp32.
// Block = 256 threads (8 waves). Tile: BM=64 x BN=128, full K staged in LDS.
// Waves arranged 2(M) x 4(N); each wave owns a 32x32 register tile:
//   2 A frags + 2 B frags -> 4 WMMAs per 32-K step (2.0 ds_load_b128 / WMMA).
// ---------------------------------------------------------------------------
template <int K, int RELU>
__global__ __launch_bounds__(256) void gemm_bias_act(
    const float* __restrict__ A, const unsigned short* __restrict__ Wt,
    const float* __restrict__ bias, float* __restrict__ C,
    int M, int N)
{
    constexpr int LDSK = K + 8;                 // row stride (16B aligned)
    constexpr int CPR  = K / 8;                 // 8-elem chunks per row
    __shared__ __align__(16) unsigned short sA[64 * LDSK];
    __shared__ __align__(16) unsigned short sB[128 * LDSK];

    const int tid  = threadIdx.x;
    const int lane = tid & 31;
    const int wave = tid >> 5;
    const int g    = lane >> 4;                 // K-half selector
    const int r    = lane & 15;                 // row (A) / col (B) within 16
    const int blockM = blockIdx.x * 64;
    const int blockN = blockIdx.y * 128;
    const int rm = (wave & 1) * 32;             // wave row base   (0 / 32)
    const int cn = (wave >> 1) * 32;            // wave col base   (0..96)
    const bool interior = (blockM + 64 <= M);

    // ---- Stage A tile (64 x K), fp32 -> bf16, b128 loads + b128 DS stores
    if (interior) {
#pragma unroll
        for (int j = 0; j < K / 32; ++j) {
            int c = tid + 256 * j;
            int row = c / CPR, off = (c % CPR) * 8;
            const float4* p = (const float4*)(A + (size_t)(blockM + row) * K + off);
            Pack8 pk = pack_f8(p[0], p[1]);
            *(uint4*)(sA + row * LDSK + off) = pk.u;
        }
    } else {
#pragma unroll
        for (int j = 0; j < K / 32; ++j) {
            int c = tid + 256 * j;
            int row = c / CPR, off = (c % CPR) * 8;
            Pack8 pk;
            if (blockM + row < M) {
                const float4* p = (const float4*)(A + (size_t)(blockM + row) * K + off);
                pk = pack_f8(p[0], p[1]);
            } else {
                pk.u = make_uint4(0u, 0u, 0u, 0u);
            }
            *(uint4*)(sA + row * LDSK + off) = pk.u;
        }
    }

    // ---- Stage Wt tile (128 x K) bf16: straight b128 copies
#pragma unroll
    for (int j = 0; j < K / 16; ++j) {
        int c = tid + 256 * j;
        int row = c / CPR, off = (c % CPR) * 8;
        uint4 v = *(const uint4*)(Wt + (size_t)(blockN + row) * K + off);
        *(uint4*)(sB + row * LDSK + off) = v;
    }
    __syncthreads();

    // ---- Register-resident K loop: 4 WMMAs per step, no barriers
    v8f acc00 = {}, acc01 = {}, acc10 = {}, acc11 = {};
    const unsigned short* ar0 = sA + (rm + r) * LDSK;
    const unsigned short* ar1 = sA + (rm + 16 + r) * LDSK;
    const unsigned short* br0 = sB + (cn + r) * LDSK;
    const unsigned short* br1 = sB + (cn + 16 + r) * LDSK;
#pragma unroll
    for (int k0 = 0; k0 < K; k0 += 32) {
        // A frags (16x32): lane-group g holds K = 8g..8g+7 and 16+8g..16+8g+7
        v16bf a0 = make_frag(ar0 + k0 + 8 * g, ar0 + k0 + 16 + 8 * g);
        v16bf a1 = make_frag(ar1 + k0 + 8 * g, ar1 + k0 + 16 + 8 * g);
        // B frags (32x16): lane holds column r, K = 8g..8g+7 and 16+8g..16+8g+7
        v16bf b0 = make_frag(br0 + k0 + 8 * g, br0 + k0 + 16 + 8 * g);
        v16bf b1 = make_frag(br1 + k0 + 8 * g, br1 + k0 + 16 + 8 * g);
        acc00 = wmma_bf16(a0, b0, acc00);   // adjacent pairs share an operand
        acc01 = wmma_bf16(a0, b1, acc01);
        acc10 = wmma_bf16(a1, b0, acc10);
        acc11 = wmma_bf16(a1, b1, acc11);
    }

    // ---- Epilogue: VGPR v holds row m = 8*(lane/16)+v, col n = lane&15
    const int c0 = blockN + cn + r;
    const int c1 = c0 + 16;
    const float bv0 = bias[c0];             // hoisted: invariant per lane
    const float bv1 = bias[c1];
    if (interior) {
#pragma unroll
        for (int v = 0; v < 8; ++v) {
            int row0 = blockM + rm + g * 8 + v;
            int row1 = row0 + 16;
            float o00 = acc00[v] + bv0, o01 = acc01[v] + bv1;
            float o10 = acc10[v] + bv0, o11 = acc11[v] + bv1;
            if (RELU) {
                o00 = fmaxf(o00, 0.0f); o01 = fmaxf(o01, 0.0f);
                o10 = fmaxf(o10, 0.0f); o11 = fmaxf(o11, 0.0f);
            }
            C[(size_t)row0 * N + c0] = o00;
            C[(size_t)row0 * N + c1] = o01;
            C[(size_t)row1 * N + c0] = o10;
            C[(size_t)row1 * N + c1] = o11;
        }
    } else {
#pragma unroll
        for (int v = 0; v < 8; ++v) {
            int row0 = blockM + rm + g * 8 + v;
            int row1 = row0 + 16;
            float o00 = acc00[v] + bv0, o01 = acc01[v] + bv1;
            float o10 = acc10[v] + bv0, o11 = acc11[v] + bv1;
            if (RELU) {
                o00 = fmaxf(o00, 0.0f); o01 = fmaxf(o01, 0.0f);
                o10 = fmaxf(o10, 0.0f); o11 = fmaxf(o11, 0.0f);
            }
            if (row0 < M) {
                C[(size_t)row0 * N + c0] = o00;
                C[(size_t)row0 * N + c1] = o01;
            }
            if (row1 < M) {
                C[(size_t)row1 * N + c0] = o10;
                C[(size_t)row1 * N + c1] = o11;
            }
        }
    }
}

// ---------------------------------------------------------------------------
// agg[dst] += feat[src]  (one wave per edge; float4 gathers, f32 atomics to L2)
// ---------------------------------------------------------------------------
template <int Ch>
__global__ __launch_bounds__(256) void scatter_add_kernel(
    const float* __restrict__ feat, const long long* __restrict__ src,
    const long long* __restrict__ dst, float* __restrict__ agg, int nEdges)
{
    const int lane = threadIdx.x & 31;
    int wid = blockIdx.x * (blockDim.x >> 5) + (threadIdx.x >> 5);
    const int nw = gridDim.x * (blockDim.x >> 5);
    for (int e = wid; e < nEdges; e += nw) {
        long long s = src[e];
        long long d = dst[e];
        const float4* fs = (const float4*)(feat + (size_t)s * Ch);
        float* ad = agg + (size_t)d * Ch;
#pragma unroll
        for (int j = 0; j < Ch / 128; ++j) {
            float4 v = fs[lane + 32 * j];
            int base = (lane + 32 * j) * 4;
            atomicAdd(ad + base + 0, v.x);
            atomicAdd(ad + base + 1, v.y);
            atomicAdd(ad + base + 2, v.z);
            atomicAdd(ad + base + 3, v.w);
        }
    }
}

// Vectorized copy: dst = src (count float4 elements)
__global__ __launch_bounds__(256) void copy_f4_kernel(
    const float4* __restrict__ src, float4* __restrict__ dst, int n4)
{
    int i = blockIdx.x * blockDim.x + threadIdx.x;
    int stride = gridDim.x * blockDim.x;
    for (; i < n4; i += stride) dst[i] = src[i];
}

// ---------------------------------------------------------------------------
// Host-side launcher
// ---------------------------------------------------------------------------
extern "C" void kernel_launch(void* const* d_in, const int* in_sizes, int n_in,
                              void* d_out, int out_size, void* d_ws, size_t ws_size,
                              hipStream_t stream) {
    const float*     x    = (const float*)d_in[0];
    const long long* ei   = (const long long*)d_in[1]; // int64 per reference
    const float*     W1a  = (const float*)d_in[2];
    const float*     b1a  = (const float*)d_in[3];
    const float*     W2a  = (const float*)d_in[4];
    const float*     b2a  = (const float*)d_in[5];
    const float*     W1b  = (const float*)d_in[6];
    const float*     b1b  = (const float*)d_in[7];
    const float*     W2b  = (const float*)d_in[8];
    const float*     b2b  = (const float*)d_in[9];
    float* out = (float*)d_out;

    const int IN_C = 128, HID = 256, OUT_C = 128;
    const int nodes = in_sizes[0] / IN_C;       // 100000
    const int edges = in_sizes[1] / 2;          // 1600000
    const long long* srcIdx = ei;
    const long long* dstIdx = ei + edges;

    // Workspace: three L2-resident node buffers + four bf16-transposed weights
    float* B0 = (float*)d_ws;                    // agg buffer
    float* B1 = B0 + (size_t)nodes * HID;        // MLP hidden
    float* B2 = B1 + (size_t)nodes * HID;        // h1 = relu(conv1)
    unsigned short* Wt1a = (unsigned short*)(B2 + (size_t)nodes * HID);
    unsigned short* Wt2a = Wt1a + 256 * 256;
    unsigned short* Wt1b = Wt2a + 256 * 256;
    unsigned short* Wt2b = Wt1b + 256 * 256;

    dim3 blk(256, 1, 1);
    const int scatterBlocks = (edges + 7) / 8;   // 8 waves/block, 1 edge/wave
    const int gemmMB = (nodes + 63) / 64;

    // ----- Weight prep (tiny) -----
    transpose_w_bf16<<<(IN_C * HID + 255) / 256, blk, 0, stream>>>(W1a, Wt1a, IN_C, HID);
    transpose_w_bf16<<<(HID * HID + 255) / 256, blk, 0, stream>>>(W2a, Wt2a, HID, HID);
    transpose_w_bf16<<<(HID * HID + 255) / 256, blk, 0, stream>>>(W1b, Wt1b, HID, HID);
    transpose_w_bf16<<<(HID * OUT_C + 255) / 256, blk, 0, stream>>>(W2b, Wt2b, HID, OUT_C);

    // ----- Layer 1: agg1 = x + segment_sum(x[src], dst) -----
    {
        int n4 = nodes * IN_C / 4;
        copy_f4_kernel<<<(n4 + 255) / 256, blk, 0, stream>>>(
            (const float4*)x, (float4*)B0, n4);
        scatter_add_kernel<128><<<scatterBlocks, blk, 0, stream>>>(
            x, srcIdx, dstIdx, B0, edges);
        // B1 = relu(B0 @ W1a + b1a)   [nodes,128]@[128,256]
        gemm_bias_act<128, 1><<<dim3(gemmMB, HID / 128), blk, 0, stream>>>(
            B0, Wt1a, b1a, B1, nodes, HID);
        // B2 = relu(B1 @ W2a + b2a)   (conv1 output + outer relu fused)
        gemm_bias_act<256, 1><<<dim3(gemmMB, HID / 128), blk, 0, stream>>>(
            B1, Wt2a, b2a, B2, nodes, HID);
    }

    // ----- Layer 2: agg2 = h1 + segment_sum(h1[src], dst) -----
    {
        int n4 = nodes * HID / 4;
        copy_f4_kernel<<<(n4 + 255) / 256, blk, 0, stream>>>(
            (const float4*)B2, (float4*)B0, n4);
        scatter_add_kernel<256><<<scatterBlocks, blk, 0, stream>>>(
            B2, srcIdx, dstIdx, B0, edges);
        // B1 = relu(B0 @ W1b + b1b)   [nodes,256]@[256,256]
        gemm_bias_act<256, 1><<<dim3(gemmMB, HID / 128), blk, 0, stream>>>(
            B0, Wt1b, b1b, B1, nodes, HID);
        // out = B1 @ W2b + b2b        [nodes,256]@[256,128], no relu
        gemm_bias_act<256, 0><<<dim3(gemmMB, OUT_C / 128), blk, 0, stream>>>(
            B1, Wt2b, b2b, out, nodes, HID);
    }
}
